// BlockAttention_69904887710028
// MI455X (gfx1250) — matrix operand order repaired and checked by reading, not compiled, not run on hardware
//
#include <hip/hip_runtime.h>

// ---------------------------------------------------------------------------
// Fused Swin-style 8x8 windowed attention block for MI455X (gfx1250, wave32).
// One workgroup (256 threads = 8 waves) per window; all intermediates in LDS.
// All WMMA operands stored in fragment-native LDS layout:
//   fragment = 32 lane-slots x 16 bf16 halves, lane's halves contiguous (32B)
//   -> v16bf load/store == 2x ds_load_b128 / ds_store_b128.
// GEMMs use v_wmma_f32_16x16x32_bf16 with fp32 accumulation.
// ---------------------------------------------------------------------------

typedef __attribute__((ext_vector_type(16))) __bf16 v16bf;
typedef __attribute__((ext_vector_type(8)))  float  v8f;

#define NWIN   2048      // 8 * 16 * 16 windows
#define CCH    256

// ---- LDS layout (bytes), all 32B aligned ----------------------------------
#define XS        68u                        // fp32 row stride for x window
#define OFF_X     0u                         // [256][68] f32 = 69632
#define OFF_TBL   69632u                     // [225][8] f32  = 7200
#define OFF_VEC   76832u                     // 4 x 64 f32    = 1024
#define OFF_XN    77856u                     // xn  Bfrag [8kt][4nt]  = 32768
#define OFF_QA    110624u                    // q   Afrag [8h][4mt]   = 32768
#define OFF_KB    143392u                    // k   Bfrag [8h][4nt]   = 32768
#define OFF_VB    176160u                    // v   Bfrag [8h][2kt][2nt] = 32768
#define OFF_PA    208928u                    // P   Afrag [8w][4mt][2kt] = 65536
#define OFF_AO    274464u                    // ao  Bfrag [8kt][4nt]  = 32768
#define SMEM_BYTES 307232u                   // < 320 KB per WGP

static __device__ __forceinline__ unsigned short f2bfu(float f) {
  return __builtin_bit_cast(unsigned short, static_cast<__bf16>(f));
}
static __device__ __forceinline__ unsigned int pk2(float a, float b) {
  return (unsigned int)f2bfu(a) | ((unsigned int)f2bfu(b) << 16);
}
// A-matrix (16x32 bf16) K index for half-element e (0..15) in lane `lane`
// (ISA 7.12.2: lanes 0-15 hold K 0-7/16-23, lanes 16-31 hold K 8-15/24-31)
static __device__ __forceinline__ int akmap(int e, int lane) {
  return ((e >> 3) << 4) + ((lane >= 16) ? 8 : 0) + (e & 7);
}
static __device__ __forceinline__ v8f wmma_bf16(v16bf a, v16bf b, v8f c) {
  return __builtin_amdgcn_wmma_f32_16x16x32_bf16(false, a, false, b,
                                                 (short)0, c, false, false);
}
static __device__ __forceinline__ v16bf ldsv(const unsigned short* p, int halfOff) {
  return *(const v16bf*)(p + halfOff);          // 32B -> 2x ds_load_b128
}
static __device__ __forceinline__ void stsv(unsigned short* p, int halfOff, v16bf v) {
  *(v16bf*)(p + halfOff) = v;                   // 32B -> 2x ds_store_b128
}
// Load a 16x32 bf16 A-fragment of a row-major fp32 matrix from global:
// per lane the needed K indices are one contiguous 16-float run (k = 16*(lane/16) + e).
static __device__ __forceinline__ v16bf load_a_global_f32(const float* row,
                                                          int lane) {
  const int k0 = (lane >= 16) ? 16 : 0;
  float4 f0 = *(const float4*)(row + k0);
  float4 f1 = *(const float4*)(row + k0 + 4);
  float4 f2 = *(const float4*)(row + k0 + 8);
  float4 f3 = *(const float4*)(row + k0 + 12);
  v16bf a;
  a[0] = (__bf16)f0.x;  a[1] = (__bf16)f0.y;  a[2]  = (__bf16)f0.z;  a[3]  = (__bf16)f0.w;
  a[4] = (__bf16)f1.x;  a[5] = (__bf16)f1.y;  a[6]  = (__bf16)f1.z;  a[7]  = (__bf16)f1.w;
  a[8] = (__bf16)f2.x;  a[9] = (__bf16)f2.y;  a[10] = (__bf16)f2.z;  a[11] = (__bf16)f2.w;
  a[12] = (__bf16)f3.x; a[13] = (__bf16)f3.y; a[14] = (__bf16)f3.z;  a[15] = (__bf16)f3.w;
  return a;
}

__global__ void __launch_bounds__(256)
swin_window_attn(const float* __restrict__ X,
                 const float* __restrict__ preg, const float* __restrict__ preb,
                 const float* __restrict__ postg, const float* __restrict__ postb,
                 const float* __restrict__ Wqkv, const float* __restrict__ bqkv,
                 const float* __restrict__ Wproj, const float* __restrict__ bproj,
                 const float* __restrict__ relTbl,
                 float* __restrict__ Y) {
  extern __shared__ unsigned char smem[];
  float*          lds_x   = (float*)(smem + OFF_X);
  float*          lds_tbl = (float*)(smem + OFF_TBL);
  float*          lds_pg  = (float*)(smem + OFF_VEC);
  float*          lds_pb  = (float*)(smem + OFF_VEC + 256);
  float*          lds_qg  = (float*)(smem + OFF_VEC + 512);
  float*          lds_qb  = (float*)(smem + OFF_VEC + 768);
  unsigned short* lds_xn  = (unsigned short*)(smem + OFF_XN);
  unsigned short* lds_qa  = (unsigned short*)(smem + OFF_QA);
  unsigned short* lds_kb  = (unsigned short*)(smem + OFF_KB);
  unsigned short* lds_vb  = (unsigned short*)(smem + OFF_VB);
  unsigned short* lds_pa  = (unsigned short*)(smem + OFF_PA);
  unsigned short* lds_ao  = (unsigned short*)(smem + OFF_AO);

  const int tid  = threadIdx.x;
  const int lane = tid & 31;
  const int w    = tid >> 5;                 // wave id == head id in phase D
  const int lm   = lane & 15;                // C-layout column
  const int lrow = (lane >= 16) ? 8 : 0;     // C-layout row offset

  const int b  = blockIdx.x;
  const int n  = b >> 8;
  const int hb = (b >> 4) & 15;
  const int wb = b & 15;

  // ---------------- Phase A: load window, LN, write xn B-fragments ---------
  {
    const float* xbase =
        X + ((((size_t)n * CCH + tid) * 128) + (size_t)hb * 8) * 128 + (size_t)wb * 8;
    float xv[64];
    float sum = 0.f, sq = 0.f;
#pragma unroll
    for (int i = 0; i < 8; ++i) {
      float4 lo = *(const float4*)(xbase + (size_t)i * 128);
      float4 hi = *(const float4*)(xbase + (size_t)i * 128 + 4);
      ((float4*)(lds_x + tid * XS + i * 8))[0] = lo;
      ((float4*)(lds_x + tid * XS + i * 8))[1] = hi;
      xv[i*8+0]=lo.x; xv[i*8+1]=lo.y; xv[i*8+2]=lo.z; xv[i*8+3]=lo.w;
      xv[i*8+4]=hi.x; xv[i*8+5]=hi.y; xv[i*8+6]=hi.z; xv[i*8+7]=hi.w;
      sum += lo.x + lo.y + lo.z + lo.w + hi.x + hi.y + hi.z + hi.w;
      sq  += lo.x*lo.x + lo.y*lo.y + lo.z*lo.z + lo.w*lo.w
           + hi.x*hi.x + hi.y*hi.y + hi.z*hi.z + hi.w*hi.w;
    }
    if (tid < 64) {
      lds_pg[tid] = preg[tid];  lds_pb[tid] = preb[tid];
      lds_qg[tid] = postg[tid]; lds_qb[tid] = postb[tid];
    }
    for (int i = tid; i < 225 * 8; i += 256) lds_tbl[i] = relTbl[i];
    __builtin_prefetch(Wproj + tid * 64, 0, 1);
    __syncthreads();

    const float mu   = sum * (1.f / 64.f);
    const float var  = sq * (1.f / 64.f) - mu * mu;
    const float rstd = rsqrtf(var + 1e-5f);
    // xn B-fragment: c==tid gives K; slot = l%16 + 16*((c%32)>=16), e = c%16
    const int kt  = tid >> 5;
    const int lsh = (tid & 16) ? 16 : 0;
    const int e   = tid & 15;
#pragma unroll 4
    for (int l = 0; l < 64; ++l) {
      float nv = (xv[l] - mu) * rstd * lds_pg[l] + lds_pb[l];
      lds_xn[((kt * 4 + (l >> 4)) * 32 + (l & 15) + lsh) * 16 + e] = f2bfu(nv);
    }
  }
  __syncthreads();

  // ---------------- Phase B: QKV GEMM  [768x256] @ [256x64] ----------------
  for (int mi = 0; mi < 6; ++mi) {
    const int mt = w * 6 + mi;                       // output tile over 3A=768
    float bq[8];
#pragma unroll
    for (int r = 0; r < 8; ++r) bq[r] = bqkv[mt * 16 + lrow + r];
    v8f acc[4];
#pragma unroll
    for (int nt = 0; nt < 4; ++nt)
#pragma unroll
      for (int r = 0; r < 8; ++r) acc[nt][r] = bq[r];

    for (int kt = 0; kt < 8; ++kt) {
      v16bf A = load_a_global_f32(
          Wqkv + (size_t)(mt * 16 + lm) * CCH + kt * 32, lane);
#pragma unroll
      for (int nt = 0; nt < 4; ++nt)
        acc[nt] = wmma_bf16(A, ldsv(lds_xn, (kt * 4 + nt) * 512 + lane * 16),
                            acc[nt]);
    }

    const int sect = mt >> 4;                        // 0=q, 1=k, 2=v
    const int ch0  = mt * 16 + lrow;
    const int h    = (ch0 >> 5) & 7;
    const int d0   = ch0 & 31;                       // in {0,8,16,24}
    if (sect == 0) {                                 // q -> Afrag, 8-half run
      const int slot = lm + ((d0 & 8) ? 16 : 0);
      const int e0   = (d0 & 16) >> 1;
#pragma unroll
      for (int nt = 0; nt < 4; ++nt) {
        uint4 pkd = {pk2(acc[nt][0], acc[nt][1]), pk2(acc[nt][2], acc[nt][3]),
                     pk2(acc[nt][4], acc[nt][5]), pk2(acc[nt][6], acc[nt][7])};
        *(uint4*)(lds_qa + (h * 4 + nt) * 512 + slot * 16 + e0) = pkd;
      }
    } else if (sect == 1) {                          // k -> Bfrag, 8-half run
      const int slot = lm + ((d0 & 8) ? 16 : 0);
      const int e0   = (d0 & 16) >> 1;
#pragma unroll
      for (int nt = 0; nt < 4; ++nt) {
        uint4 pkd = {pk2(acc[nt][0], acc[nt][1]), pk2(acc[nt][2], acc[nt][3]),
                     pk2(acc[nt][4], acc[nt][5]), pk2(acc[nt][6], acc[nt][7])};
        *(uint4*)(lds_kb + (h * 4 + nt) * 512 + slot * 16 + e0) = pkd;
      }
    } else {                                         // v -> Bfrag, scattered
#pragma unroll
      for (int nt = 0; nt < 4; ++nt) {
        const int fi = (h * 2 + (nt >> 1)) * 2 + (d0 >> 4);
        const int sb = ((nt & 1) ? 16 : 0);
#pragma unroll
        for (int r = 0; r < 8; ++r)
          lds_vb[(fi * 32 + ((d0 & 15) + r) + sb) * 16 + lm] = f2bfu(acc[nt][r]);
      }
    }
  }
  __syncthreads();

  // ---------------- Phase C: L2-normalize q,k (32B-chunk vectorized) -------
#pragma unroll
  for (int t = 0; t < 4; ++t) {
    const int task = tid + t * 256;          // 1024 tasks: {q,k} x head x token
    unsigned short* sec = (task & 512) ? lds_kb : lds_qa;
    const int off = ((((task >> 6) & 7) * 4 + ((task >> 4) & 3)) * 32 +
                     (task & 15)) * 16;
    v16bf c0 = ldsv(sec, off);
    v16bf c1 = ldsv(sec, off + 256);
    float ss = 0.f;
#pragma unroll
    for (int e = 0; e < 16; ++e) {
      float a = (float)c0[e], bb = (float)c1[e];
      ss += a * a + bb * bb;
    }
    const float inv = 1.f / fmaxf(sqrtf(ss), 1e-12f);
#pragma unroll
    for (int e = 0; e < 16; ++e) {
      c0[e] = (__bf16)((float)c0[e] * inv);
      c1[e] = (__bf16)((float)c1[e] * inv);
    }
    stsv(sec, off, c0);
    stsv(sec, off + 256, c1);
  }
  __syncthreads();

  // ---------------- Phase D: per-head attention (wave == head) -------------
  {
    const int h = w;
    unsigned short* pst = lds_pa + w * 4096;

    // S = q k^T + bias   (M=64, N=64, K=32 -> 16 WMMA)
    v8f s[4][4];
#pragma unroll
    for (int mt = 0; mt < 4; ++mt)
#pragma unroll
      for (int nt = 0; nt < 4; ++nt)
#pragma unroll
        for (int r = 0; r < 8; ++r) {
          int row = mt * 16 + lrow + r, col = nt * 16 + lm;
          int idx = ((row >> 3) - (col >> 3) + 7) * 15 + ((row & 7) - (col & 7) + 7);
          s[mt][nt][r] = lds_tbl[idx * 8 + h];
        }
    v16bf KB[4];
#pragma unroll
    for (int nt = 0; nt < 4; ++nt)
      KB[nt] = ldsv(lds_kb, (h * 4 + nt) * 512 + lane * 16);
#pragma unroll
    for (int mt = 0; mt < 4; ++mt) {
      v16bf A = ldsv(lds_qa, (h * 4 + mt) * 512 + lane * 16);
#pragma unroll
      for (int nt = 0; nt < 4; ++nt) s[mt][nt] = wmma_bf16(A, KB[nt], s[mt][nt]);
    }

    // Row softmax: each row lives in one 16-lane half across the 4 N-tiles.
#pragma unroll
    for (int mt = 0; mt < 4; ++mt)
#pragma unroll
      for (int r = 0; r < 8; ++r) {
        float mx = fmaxf(fmaxf(s[mt][0][r], s[mt][1][r]),
                         fmaxf(s[mt][2][r], s[mt][3][r]));
        mx = fmaxf(mx, __shfl_xor(mx, 8, 32));
        mx = fmaxf(mx, __shfl_xor(mx, 4, 32));
        mx = fmaxf(mx, __shfl_xor(mx, 2, 32));
        mx = fmaxf(mx, __shfl_xor(mx, 1, 32));
        float e0 = __expf(s[mt][0][r] - mx), e1 = __expf(s[mt][1][r] - mx);
        float e2 = __expf(s[mt][2][r] - mx), e3 = __expf(s[mt][3][r] - mx);
        float sm = e0 + e1 + e2 + e3;
        sm += __shfl_xor(sm, 8, 32);
        sm += __shfl_xor(sm, 4, 32);
        sm += __shfl_xor(sm, 2, 32);
        sm += __shfl_xor(sm, 1, 32);
        float inv = 1.f / sm;
        s[mt][0][r] = e0 * inv; s[mt][1][r] = e1 * inv;
        s[mt][2][r] = e2 * inv; s[mt][3][r] = e3 * inv;
      }

    // Stage P into its A-fragment layout (C-layout rows scatter over slots).
#pragma unroll
    for (int mt = 0; mt < 4; ++mt)
#pragma unroll
      for (int nt = 0; nt < 4; ++nt) {
        const int base = (mt * 2 + (nt >> 1)) * 512 + ((nt & 1) << 8) + lm;
#pragma unroll
        for (int r = 0; r < 8; ++r)
          pst[base + (lrow + r) * 16] = f2bfu(s[mt][nt][r]);
      }
    // producer == consumer wave: wait LDS stores, no workgroup barrier needed
    asm volatile("s_wait_dscnt 0" ::: "memory");

    // O = P V   (M=64, N=32, K=64 -> 16 WMMA)
    v16bf VB[2][2];
#pragma unroll
    for (int kt = 0; kt < 2; ++kt)
#pragma unroll
      for (int nt = 0; nt < 2; ++nt)
        VB[kt][nt] = ldsv(lds_vb, ((h * 2 + kt) * 2 + nt) * 512 + lane * 16);
#pragma unroll
    for (int mt = 0; mt < 4; ++mt) {
      v16bf PA0 = ldsv(pst, (mt * 2 + 0) * 512 + lane * 16);
      v16bf PA1 = ldsv(pst, (mt * 2 + 1) * 512 + lane * 16);
#pragma unroll
      for (int nt = 0; nt < 2; ++nt) {
        v8f o;
#pragma unroll
        for (int r = 0; r < 8; ++r) o[r] = 0.f;
        o = wmma_bf16(PA0, VB[0][nt], o);
        o = wmma_bf16(PA1, VB[1][nt], o);
        // ao B-fragment: slot = token%16 + 16*nt, e = lm  (scattered over r)
#pragma unroll
        for (int r = 0; r < 8; ++r)
          lds_ao[((h * 4 + mt) * 32 + (lrow + r) + nt * 16) * 16 + lm] =
              f2bfu(o[r]);
      }
    }
  }
  __syncthreads();

  // -------- Phase E: projection + residual + post-LN + store ---------------
  for (int mi = 0; mi < 2; ++mi) {
    const int mt = w * 2 + mi;
    float bp[8];
#pragma unroll
    for (int r = 0; r < 8; ++r) bp[r] = bproj[mt * 16 + lrow + r];
    v8f acc[4];
#pragma unroll
    for (int nt = 0; nt < 4; ++nt)
#pragma unroll
      for (int r = 0; r < 8; ++r) acc[nt][r] = bp[r];

    for (int kt = 0; kt < 8; ++kt) {
      v16bf A = load_a_global_f32(
          Wproj + (size_t)(mt * 16 + lm) * CCH + kt * 32, lane);
#pragma unroll
      for (int nt = 0; nt < 4; ++nt)
        acc[nt] = wmma_bf16(A, ldsv(lds_ao, (kt * 4 + nt) * 512 + lane * 16),
                            acc[nt]);
    }

    // y = x + attn_proj ; out = y + LN(y)  (LN over the 64 tokens of a row)
    float y[4][8];
#pragma unroll
    for (int nt = 0; nt < 4; ++nt)
#pragma unroll
      for (int r = 0; r < 8; ++r)
        y[nt][r] = acc[nt][r] + lds_x[(mt * 16 + lrow + r) * XS + nt * 16 + lm];

#pragma unroll
    for (int r = 0; r < 8; ++r) {
      float sm = y[0][r] + y[1][r] + y[2][r] + y[3][r];
      float sq = y[0][r]*y[0][r] + y[1][r]*y[1][r] + y[2][r]*y[2][r] + y[3][r]*y[3][r];
      sm += __shfl_xor(sm, 8, 32); sq += __shfl_xor(sq, 8, 32);
      sm += __shfl_xor(sm, 4, 32); sq += __shfl_xor(sq, 4, 32);
      sm += __shfl_xor(sm, 2, 32); sq += __shfl_xor(sq, 2, 32);
      sm += __shfl_xor(sm, 1, 32); sq += __shfl_xor(sq, 1, 32);
      const float mu   = sm * (1.f / 64.f);
      const float var  = sq * (1.f / 64.f) - mu * mu;
      const float rstd = rsqrtf(var + 1e-5f);
      const int   c    = mt * 16 + lrow + r;
#pragma unroll
      for (int nt = 0; nt < 4; ++nt) {
        const int   l = nt * 16 + lm;
        const float o = y[nt][r] + (y[nt][r] - mu) * rstd * lds_qg[l] + lds_qb[l];
        Y[((((size_t)n * CCH + c) * 128) + (size_t)hb * 8 + (l >> 3)) * 128 +
          (size_t)wb * 8 + (l & 7)] = o;
      }
    }
  }
}

extern "C" void kernel_launch(void* const* d_in, const int* in_sizes, int n_in,
                              void* d_out, int out_size, void* d_ws, size_t ws_size,
                              hipStream_t stream) {
  (void)in_sizes; (void)n_in; (void)out_size; (void)d_ws; (void)ws_size;
  const float* X     = (const float*)d_in[0];
  const float* preg  = (const float*)d_in[1];
  const float* preb  = (const float*)d_in[2];
  const float* postg = (const float*)d_in[3];
  const float* postb = (const float*)d_in[4];
  const float* Wqkv  = (const float*)d_in[5];
  const float* bqkv  = (const float*)d_in[6];
  const float* Wproj = (const float*)d_in[7];
  const float* bproj = (const float*)d_in[8];
  const float* tbl   = (const float*)d_in[9];
  float* Y = (float*)d_out;

  swin_window_attn<<<dim3(NWIN), dim3(256), SMEM_BYTES, stream>>>(
      X, preg, preb, postg, postb, Wqkv, bqkv, Wproj, bproj, tbl, Y);
}
